// ConcatLayer_48627619726065
// MI455X (gfx1250) — compile-verified, matrix operand order checked by reading, not run on hardware
//
#include <hip/hip_runtime.h>

typedef __attribute__((ext_vector_type(16))) _Float16 v16h;
typedef __attribute__((ext_vector_type(8)))  float    v8f;

#define N_BONDS_C        50000
#define N_EDGES_C        600000
#define FEAT             128      // BOND_DIM + SPH_DIM
#define UNITS_C          64
#define SEGS_PER_BLOCK   16
#define THREADS_PER_BLK  128      // 4 wave32s

// lower_bound over the sorted segment column (edges_neighbor[:,0], stride 2)
__device__ __forceinline__ int lbound_seg(const int* __restrict__ en, int val) {
    int lo = 0, hi = N_EDGES_C;
    while (lo < hi) {
        int mid = (lo + hi) >> 1;
        if (en[2 * mid] < val) lo = mid + 1; else hi = mid;
    }
    return lo;
}

__global__ __launch_bounds__(THREADS_PER_BLK)
void fused_concat_segsum_gemm(const float* __restrict__ bond,   // 50000 x 64
                              const float* __restrict__ sph,    // 600000 x 64
                              const int*   __restrict__ en,     // 600000 x 2 [seg, nbr]
                              const float* __restrict__ W,      // 128 x 64
                              const float* __restrict__ bias,   // 64
                              float*       __restrict__ out)    // 50000 x 64
{
    __shared__ __align__(16) float    aggf32[SEGS_PER_BLOCK * FEAT]; // 8 KB
    __shared__ __align__(16) _Float16 aggf16[SEGS_PER_BLOCK * FEAT]; // 4 KB
    __shared__ __align__(16) _Float16 wT[UNITS_C * FEAT];            // 16 KB, [n][k]

    const int tid  = threadIdx.x;
    const int wave = tid >> 5;
    const int lane = tid & 31;
    const int seg0 = blockIdx.x * SEGS_PER_BLOCK;

    // ---- init LDS accumulator + stage weights transposed as f16 ----
    for (int i = tid; i < SEGS_PER_BLOCK * FEAT; i += THREADS_PER_BLK)
        aggf32[i] = 0.0f;
    for (int i = tid; i < FEAT * UNITS_C; i += THREADS_PER_BLK) {
        int k = i >> 6, n = i & 63;            // W is row-major [k][n]
        wT[n * FEAT + k] = (_Float16)W[i];     // store [n][k] for contiguous K loads
    }
    __syncthreads();

    // ---- edge range for this block's 16 segments (seg column is sorted) ----
    const int lo = lbound_seg(en, seg0);
    const int hi = lbound_seg(en, seg0 + SEGS_PER_BLOCK);

    const int total = hi - lo;
    const int per   = (total + 3) >> 2;        // contiguous chunk per wave
    const int wbeg  = lo + wave * per;
    int       wend  = wbeg + per; if (wend > hi) wend = hi;

    // ---- gather + concat + segment-sum into LDS ----
    // lane owns features [4*lane, 4*lane+4): lanes 0-15 -> gathered bond row,
    // lanes 16-31 -> sph row. Register-accumulate runs of equal segment id
    // (sorted!) and flush with ds_add_f32 only on segment change.
    const int f4 = lane * 4;
    float a0 = 0.f, a1 = 0.f, a2 = 0.f, a3 = 0.f;
    int curRow = -1;
    for (int e = wbeg; e < wend; ++e) {
        const int s   = en[2 * e];
        const int row = s - seg0;
        if (row != curRow) {
            if (curRow >= 0) {
                float* p = &aggf32[curRow * FEAT + f4];
                atomicAdd(p + 0, a0); atomicAdd(p + 1, a1);
                atomicAdd(p + 2, a2); atomicAdd(p + 3, a3);
            }
            a0 = a1 = a2 = a3 = 0.f;
            curRow = row;
        }
        const float* src;
        if (lane < 16) {
            const int nb = en[2 * e + 1];
            src = bond + (size_t)nb * 64 + f4;
        } else {
            src = sph + (size_t)e * 64 + (f4 - 64);
        }
        const float4 v = *(const float4*)src;   // global_load_b128
        a0 += v.x; a1 += v.y; a2 += v.z; a3 += v.w;
    }
    if (curRow >= 0) {
        float* p = &aggf32[curRow * FEAT + f4];
        atomicAdd(p + 0, a0); atomicAdd(p + 1, a1);
        atomicAdd(p + 2, a2); atomicAdd(p + 3, a3);
    }
    __syncthreads();

    // ---- convert aggregated tile to f16 for WMMA ----
    for (int i = tid; i < SEGS_PER_BLOCK * FEAT; i += THREADS_PER_BLK)
        aggf16[i] = (_Float16)aggf32[i];
    __syncthreads();

    // ---- (16 x 128) @ (128 x 64) via 4 K-steps of v_wmma_f32_16x16x32_f16;
    //      wave w owns output columns [16w, 16w+16) ----
    // 16-bit A layout: lane L -> row M = L%16; halves 0-7 = K base+0..7,
    // halves 8-15 = K base2+0..7 with base = (L<16 ? 0 : 8), base2 = base+16.
    const int m  = lane & 15;
    const int hh = lane >> 4;
    const int n  = wave * 16 + m;
    v8f c = {};
    #pragma unroll
    for (int ks = 0; ks < 4; ++ks) {
        const int kA0 = ks * 32 + hh * 8;
        const int kA1 = ks * 32 + 16 + hh * 8;
        union { float4 q[2]; v16h v; } ua, ub;
        ua.q[0] = *(const float4*)&aggf16[m * FEAT + kA0];   // ds_load_b128
        ua.q[1] = *(const float4*)&aggf16[m * FEAT + kA1];
        ub.q[0] = *(const float4*)&wT[n * FEAT + kA0];
        ub.q[1] = *(const float4*)&wT[n * FEAT + kA1];
        c = __builtin_amdgcn_wmma_f32_16x16x32_f16(
                /*neg_a=*/false, ua.v, /*neg_b=*/false, ub.v,
                /*c_mod=*/(short)0, c, /*reuse_a=*/false, /*reuse_b=*/false);
    }

    // ---- epilogue: bias + store. C/D layout: VGPR r -> row r + 8*hh, col n.
    const float bv = bias[n];
    #pragma unroll
    for (int r = 0; r < 8; ++r) {
        const int gm = seg0 + r + hh * 8;
        out[(size_t)gm * UNITS_C + n] = c[r] + bv;
    }
}

extern "C" void kernel_launch(void* const* d_in, const int* in_sizes, int n_in,
                              void* d_out, int out_size, void* d_ws, size_t ws_size,
                              hipStream_t stream) {
    (void)in_sizes; (void)n_in; (void)out_size; (void)d_ws; (void)ws_size;
    const float* bond = (const float*)d_in[0];   // bond_features
    const float* sph  = (const float*)d_in[1];   // edges_sph_features
    const int*   en   = (const int*)  d_in[2];   // edges_neighbor (N_EDGES x 2)
    const float* W    = (const float*)d_in[3];   // kernel
    const float* bias = (const float*)d_in[4];   // bias
    float*       out  = (float*)d_out;           // 50000 x 64

    dim3 grid(N_BONDS_C / SEGS_PER_BLOCK);       // 3125 blocks
    fused_concat_segsum_gemm<<<grid, THREADS_PER_BLK, 0, stream>>>(
        bond, sph, en, W, bias, out);
}
